// RepeatedLayersSharedAll_38079180046780
// MI455X (gfx1250) — compile-verified, matrix operand order checked by reading, not run on hardware
//
#include <hip/hip_runtime.h>
#include <hip/hip_bf16.h>

// ---------------------------------------------------------------------------
// u_{l+1} = tanh(scalar * (w * (u_l @ Wh^T) + drive)),  drive = a*(x@Wi^T)+b*bias
// B=4096, H=1024, 20 layers. Compute-bound f16 WMMA GEMM (f32 accumulate),
// all operands L2-resident on MI455X (192 MB L2, 23.3 TB/s HBM irrelevant here).
// Double-buffered LDS pipeline: global loads for tile k+1 issue under the
// 8 WMMAs of tile k; one workgroup barrier per K-step.
// ---------------------------------------------------------------------------

typedef __attribute__((ext_vector_type(16))) _Float16 v16h;
typedef __attribute__((ext_vector_type(8)))  _Float16 v8h;
typedef __attribute__((ext_vector_type(8)))  float    v8f;

#define BATCH   4096
#define HID     1024
#define NLAYERS 20

#define BM 128           // block tile M
#define BN 128           // block tile N
#define BK 32            // K step = WMMA K for f16
#define LDS_STRIDE 40    // halves per LDS row (32 data + 8 pad -> conflict-free b128 reads)

__global__ __launch_bounds__(256)
void cvt_f32_to_f16(const float* __restrict__ src, _Float16* __restrict__ dst, int n) {
    int i = blockIdx.x * blockDim.x + threadIdx.x;
    if (i < n) dst[i] = (_Float16)src[i];
}

__device__ __forceinline__ float fast_tanh(float x) {
    // tanh(x) = 1 - 2/(e^{2x}+1): v_exp_f32 + v_rcp_f32, no IEEE div sequence.
    x = fminf(15.0f, fmaxf(-15.0f, x));
    float e = __expf(2.0f * x);
    return 1.0f - 2.0f * __builtin_amdgcn_rcpf(e + 1.0f);
}

// mode 0 (drive):  out32 = s0*acc + s1*bias[col]
// mode 1 (layer):  val = tanh(s1*(s0*acc + drive[row,col])); out16 = val; out32?=val
__global__ __launch_bounds__(256)
void rnn_gemm_wmma(const _Float16* __restrict__ A,      // [BATCH][HID] f16
                   const _Float16* __restrict__ W,      // [HID][HID] f16, B = W^T
                   const float*    __restrict__ addend, // mode0: bias[HID]; mode1: drive[BATCH][HID]
                   const float*    __restrict__ s0,     // mode0: a ; mode1: w
                   const float*    __restrict__ s1,     // mode0: b ; mode1: scalar
                   _Float16*       __restrict__ out16,  // mode1: next u (f16)
                   float*          __restrict__ out32,  // mode0: drive ; mode1: final output or null
                   int mode)
{
    __shared__ __align__(16) _Float16 ldsA[2][BM * LDS_STRIDE];
    __shared__ __align__(16) _Float16 ldsB[2][BN * LDS_STRIDE];

    const int tid  = threadIdx.x;
    const int lane = tid & 31;
    const int wv   = tid >> 5;      // 8 waves
    const int wvM  = wv & 3;        // 4 waves along M  -> 32 rows each
    const int wvN  = wv >> 2;       // 2 waves along N  -> 64 cols each
    const int half = lane >> 4;     // lane group (ISA f16 fragment layout)
    const int r    = lane & 15;

    const int M0 = blockIdx.x * BM;
    const int N0 = blockIdx.y * BN;

    // Per-thread staging: 2x 16B chunks of A tile, 2x 16B of B tile per K-step.
    // chunk c (0..511): row = c>>2 (0..127), kc = c&3 (16B slice of the 64B row).
    const int c0row = tid >> 2,          c0kc = tid & 3;
    const int c1row = (tid + 256) >> 2,  c1kc = tid & 3;   // (tid+256)&3 == tid&3

    v8f acc[2][4] = {};
    v8h rA0, rA1, rB0, rB1;

    // ---- prologue: load tile k0=0 and stage into buffer 0 ----
    {
        rA0 = *(const v8h*)&A[(size_t)(M0 + c0row) * HID + c0kc * 8];
        rA1 = *(const v8h*)&A[(size_t)(M0 + c1row) * HID + c1kc * 8];
        rB0 = *(const v8h*)&W[(size_t)(N0 + c0row) * HID + c0kc * 8];
        rB1 = *(const v8h*)&W[(size_t)(N0 + c1row) * HID + c1kc * 8];
        *(v8h*)&ldsA[0][c0row * LDS_STRIDE + c0kc * 8] = rA0;
        *(v8h*)&ldsA[0][c1row * LDS_STRIDE + c1kc * 8] = rA1;
        *(v8h*)&ldsB[0][c0row * LDS_STRIDE + c0kc * 8] = rB0;
        *(v8h*)&ldsB[0][c1row * LDS_STRIDE + c1kc * 8] = rB1;
    }

    int cur = 0;
    for (int k0 = 0; k0 < HID; k0 += BK) {
        __syncthreads();   // staged buffer `cur` visible; prior reads of `nxt` done
        const int nxt = cur ^ 1;
        const bool have_next = (k0 + BK) < HID;

        // Issue next tile's global loads first (L2 hits overlap the WMMAs below).
        if (have_next) {
            const int kn = k0 + BK;
            rA0 = *(const v8h*)&A[(size_t)(M0 + c0row) * HID + kn + c0kc * 8];
            rA1 = *(const v8h*)&A[(size_t)(M0 + c1row) * HID + kn + c1kc * 8];
            rB0 = *(const v8h*)&W[(size_t)(N0 + c0row) * HID + kn + c0kc * 8];
            rB1 = *(const v8h*)&W[(size_t)(N0 + c1row) * HID + kn + c1kc * 8];
        }

        // Fragments per ISA layout:
        //  lanes 0-15 hold K {0..7,16..23}, lanes 16-31 hold K {8..15,24..31}
        v16h afrag[2], bfrag[4];
        #pragma unroll
        for (int mi = 0; mi < 2; ++mi) {
            const _Float16* p = &ldsA[cur][(32 * wvM + 16 * mi + r) * LDS_STRIDE + 8 * half];
            v8h lo = *(const v8h*)p;
            v8h hi = *(const v8h*)(p + 16);
            #pragma unroll
            for (int i = 0; i < 8; ++i) { afrag[mi][i] = lo[i]; afrag[mi][8 + i] = hi[i]; }
        }
        #pragma unroll
        for (int ni = 0; ni < 4; ++ni) {
            const _Float16* p = &ldsB[cur][(64 * wvN + 16 * ni + r) * LDS_STRIDE + 8 * half];
            v8h lo = *(const v8h*)p;
            v8h hi = *(const v8h*)(p + 16);
            #pragma unroll
            for (int i = 0; i < 8; ++i) { bfrag[ni][i] = lo[i]; bfrag[ni][8 + i] = hi[i]; }
        }

        #pragma unroll
        for (int mi = 0; mi < 2; ++mi)
            #pragma unroll
            for (int ni = 0; ni < 4; ++ni)
                acc[mi][ni] = __builtin_amdgcn_wmma_f32_16x16x32_f16(
                    /*neg_a=*/false, afrag[mi],
                    /*neg_b=*/false, bfrag[ni],
                    /*c_mod=*/(short)0, acc[mi][ni],
                    /*reuse_a=*/false, /*reuse_b=*/false);

        // Stage next tile into the other buffer (current readers unaffected).
        if (have_next) {
            *(v8h*)&ldsA[nxt][c0row * LDS_STRIDE + c0kc * 8] = rA0;
            *(v8h*)&ldsA[nxt][c1row * LDS_STRIDE + c1kc * 8] = rA1;
            *(v8h*)&ldsB[nxt][c0row * LDS_STRIDE + c0kc * 8] = rB0;
            *(v8h*)&ldsB[nxt][c1row * LDS_STRIDE + c1kc * 8] = rB1;
        }
        cur = nxt;
    }

    const float sa = *s0;
    const float sb = *s1;

    // C/D layout: lanes 0-15: M=v, N=lane; lanes 16-31: M=8+v, N=lane-16.
    #pragma unroll
    for (int mi = 0; mi < 2; ++mi) {
        #pragma unroll
        for (int ni = 0; ni < 4; ++ni) {
            const int gcol = N0 + 64 * wvN + 16 * ni + r;
            #pragma unroll
            for (int v = 0; v < 8; ++v) {
                const int grow = M0 + 32 * wvM + 16 * mi + 8 * half + v;
                const size_t idx = (size_t)grow * HID + gcol;
                const float av = acc[mi][ni][v];
                if (mode == 0) {
                    out32[idx] = sa * av + sb * addend[gcol];
                } else {
                    float val = fast_tanh(sb * (sa * av + addend[idx]));
                    out16[idx] = (_Float16)val;
                    if (out32) out32[idx] = val;
                }
            }
        }
    }
}

extern "C" void kernel_launch(void* const* d_in, const int* in_sizes, int n_in,
                              void* d_out, int out_size, void* d_ws, size_t ws_size,
                              hipStream_t stream) {
    const float* initial_hidden = (const float*)d_in[0];   // [4096,1024]
    const float* x              = (const float*)d_in[1];   // [4096,1024]
    const float* hiddenWeight   = (const float*)d_in[2];   // [1024,1024]
    const float* inputWeight    = (const float*)d_in[3];   // [1024,1024]
    const float* bias           = (const float*)d_in[4];   // [1024]
    const float* scalar_p       = (const float*)d_in[5];
    const float* w_p            = (const float*)d_in[6];
    const float* b_p            = (const float*)d_in[7];
    const float* a_p            = (const float*)d_in[8];

    char* ws = (char*)d_ws;
    float*    drive = (float*)ws;                                  // 16 MB
    _Float16* W16   = (_Float16*)(ws + ((size_t)16 << 20));        //  2 MB
    _Float16* Wi16  = (_Float16*)(ws + ((size_t)18 << 20));        //  2 MB
    _Float16* x16   = (_Float16*)(ws + ((size_t)20 << 20));        //  8 MB
    _Float16* uA    = (_Float16*)(ws + ((size_t)28 << 20));        //  8 MB
    _Float16* uB    = (_Float16*)(ws + ((size_t)36 << 20));        //  8 MB (44 MB total)

    const int nBH = BATCH * HID;    // 4M
    const int nHH = HID * HID;      // 1M

    cvt_f32_to_f16<<<(nHH + 255) / 256, 256, 0, stream>>>(hiddenWeight, W16, nHH);
    cvt_f32_to_f16<<<(nHH + 255) / 256, 256, 0, stream>>>(inputWeight, Wi16, nHH);
    cvt_f32_to_f16<<<(nBH + 255) / 256, 256, 0, stream>>>(x, x16, nBH);
    cvt_f32_to_f16<<<(nBH + 255) / 256, 256, 0, stream>>>(initial_hidden, uA, nBH);

    dim3 grid(BATCH / BM, HID / BN);   // 32 x 8 blocks
    dim3 block(256);

    // drive = a*(x @ Wi^T) + b*bias   (Wi happens to be identity, but computed generally)
    rnn_gemm_wmma<<<grid, block, 0, stream>>>(x16, Wi16, bias, a_p, b_p,
                                              (_Float16*)nullptr, drive, 0);

    _Float16* uc = uA;
    _Float16* un = uB;
    for (int l = 0; l < NLAYERS; ++l) {
        float* o32 = (l == NLAYERS - 1) ? (float*)d_out : nullptr;
        rnn_gemm_wmma<<<grid, block, 0, stream>>>(uc, W16, drive, w_p, scalar_p,
                                                  un, o32, 1);
        _Float16* t = uc; uc = un; un = t;
    }
}
